// ResGCN_53197464928873
// MI455X (gfx1250) — compile-verified, matrix operand order checked by reading, not run on hardware
//
#include <hip/hip_runtime.h>
#include <hip/hip_bf16.h>
#include <math.h>

typedef __attribute__((ext_vector_type(16))) __bf16 v16bf;
typedef __attribute__((ext_vector_type(8)))  float  v8f;

#define N_NODES 646
#define FDIM    32768
#define HDIM    64
#define LDIM    128
#define NCOMB   128                 // 64 cols (x@W1) + 64 cols (x@LPw)
#define KSPLIT  32
#define KCHUNK  (FDIM / KSPLIT)     // 1024
#define MGROUPS 21                  // ceil(646/32): 2 M-tiles (32 rows) per wave
#define NTILES  8                   // 128/16
#define NL      (N_NODES * NCOMB)   // 82688
#define NH      (N_NODES * HDIM)    // 41344

// ---- fp32 -> bf16 round-to-nearest-even (scalar, cold paths) ----
__device__ __forceinline__ unsigned short f2bf(float f) {
    union { float f; unsigned u; } v; v.f = f;
    unsigned r = v.u + 0x7FFFu + ((v.u >> 16) & 1u);
    return (unsigned short)(r >> 16);
}

// ---- packed hardware convert: dword = { bf16(a) , bf16(b) } ----
__device__ __forceinline__ unsigned cvt_pk_bf16(float a, float b) {
    unsigned r;
    asm("v_cvt_pk_bf16_f32 %0, %1, %2" : "=v"(r) : "v"(a), "v"(b));
    return r;
}

// ---- build A fragment (16x32 bf16, ISA 16-bit A layout) from fp32 row ----
// p already includes k0 + half*8; reads K-runs [0..7] and [16..23].
__device__ __forceinline__ v16bf load_a_bf16(const float* __restrict__ p) {
    float4 p0 = *(const float4*)(p);
    float4 p1 = *(const float4*)(p + 4);
    float4 p2 = *(const float4*)(p + 16);
    float4 p3 = *(const float4*)(p + 20);
    union { v16bf v; unsigned d[8]; } A;
    A.d[0] = cvt_pk_bf16(p0.x, p0.y);
    A.d[1] = cvt_pk_bf16(p0.z, p0.w);
    A.d[2] = cvt_pk_bf16(p1.x, p1.y);
    A.d[3] = cvt_pk_bf16(p1.z, p1.w);
    A.d[4] = cvt_pk_bf16(p2.x, p2.y);
    A.d[5] = cvt_pk_bf16(p2.z, p2.w);
    A.d[6] = cvt_pk_bf16(p3.x, p3.y);
    A.d[7] = cvt_pk_bf16(p3.z, p3.w);
    return A.v;
}

// =====================================================================
// One-time weight transpose: src fp32 [K, N] row-major -> dst bf16 [N][dstStride]
// LDS-tiled 64x64, coalesced reads, vector bf16 writes. K,N multiples of 64.
// =====================================================================
__global__ __launch_bounds__(256)
void transpose_bf16_kernel(const float* __restrict__ src, unsigned short* __restrict__ dst,
                           int N, int dstStride) {
    __shared__ unsigned short tile[64][72];          // [n][k], padded row stride
    const int k0 = blockIdx.x * 64;
    const int n0 = blockIdx.y * 64;
    const int t  = threadIdx.x;
    const int rb = t >> 4;                           // 0..15
    const int c4 = (t & 15) * 4;                     // 0..60
#pragma unroll
    for (int i = 0; i < 4; ++i) {
        int r = rb + i * 16;                         // k within tile
        float4 w = *(const float4*)(src + (size_t)(k0 + r) * N + n0 + c4);
        tile[c4 + 0][r] = f2bf(w.x);
        tile[c4 + 1][r] = f2bf(w.y);
        tile[c4 + 2][r] = f2bf(w.z);
        tile[c4 + 3][r] = f2bf(w.w);
    }
    __syncthreads();
    const int n  = t & 63;
    const int kg = (t >> 6) * 16;                    // 0,16,32,48
    uint4 val = *(const uint4*)&tile[n][kg];
    *(uint4*)(dst + (size_t)(n0 + n) * dstStride + k0 + kg) = val;
}

// =====================================================================
// Init: zero h1acc / sums, deg=1 (self loop), out = b2 broadcast
// =====================================================================
__global__ __launch_bounds__(256)
void init_kernel(float* __restrict__ h1acc, float* __restrict__ deg,
                 float* __restrict__ sums, float* __restrict__ out,
                 const float* __restrict__ b2) {
    int i = blockIdx.x * blockDim.x + threadIdx.x;
    if (i < NL)      out[i]   = b2[i & (LDIM - 1)];
    if (i < NH)      h1acc[i] = 0.0f;
    if (i < N_NODES) deg[i]   = 1.0f;                // self loop contribution
    if (i < 2)       sums[i]  = 0.0f;
}

// =====================================================================
// GEMM1: partial[ks][646,128] = x @ [W1 | LPw]  (bf16 WMMA, split-K slices)
// grid = (MGROUPS, KSPLIT), block = 32 (one wave); 2 M-tiles per wave.
// B fragments load straight from pre-transposed bf16 Wt (L2-resident).
// =====================================================================
__global__ __launch_bounds__(32)
void gemm1_wmma(const float* __restrict__ x,
                const unsigned short* __restrict__ wt,     // [128][FDIM] bf16
                float* __restrict__ partial) {
    const int lane = threadIdx.x;
    const int half = lane >> 4;
    const int l15  = lane & 15;
    const int kk   = half * 8;
    const int m0   = blockIdx.x * 32;
    const int kbase = blockIdx.y * KCHUNK;

    int mr0 = m0 + l15;      if (mr0 > N_NODES - 1) mr0 = N_NODES - 1;
    int mr1 = m0 + 16 + l15; if (mr1 > N_NODES - 1) mr1 = N_NODES - 1;
    const float* __restrict__ xrow0 = x + (size_t)mr0 * FDIM;
    const float* __restrict__ xrow1 = x + (size_t)mr1 * FDIM;
    const unsigned short* __restrict__ brow = wt + (size_t)l15 * FDIM;

    v8f zero = {};
    v8f acc0[NTILES], acc1[NTILES];
#pragma unroll
    for (int t = 0; t < NTILES; ++t) { acc0[t] = zero; acc1[t] = zero; }

    for (int k0 = kbase; k0 < kbase + KCHUNK; k0 += 32) {
        v16bf A0 = load_a_bf16(xrow0 + k0 + kk);
        v16bf A1 = load_a_bf16(xrow1 + k0 + kk);
        __builtin_prefetch(xrow0 + k0 + 256, 0, 1);      // global_prefetch_b8
        __builtin_prefetch(xrow1 + k0 + 256, 0, 1);
#pragma unroll
        for (int t = 0; t < NTILES; ++t) {
            union { v16bf v; uint4 q[2]; } B;
            const unsigned short* p = brow + (size_t)(t * 16) * FDIM + k0 + kk;
            B.q[0] = *(const uint4*)(p);
            B.q[1] = *(const uint4*)(p + 16);
            acc0[t] = __builtin_amdgcn_wmma_f32_16x16x32_bf16(
                false, A0, false, B.v, (short)0, acc0[t], false, false);
            acc1[t] = __builtin_amdgcn_wmma_f32_16x16x32_bf16(
                false, A1, false, B.v, (short)0, acc1[t], false, false);
        }
    }

    float* __restrict__ ps = partial + (size_t)blockIdx.y * NL;
#pragma unroll
    for (int t = 0; t < NTILES; ++t) {
#pragma unroll
        for (int j = 0; j < 8; ++j) {
            const int n  = t * 16 + l15;
            const int ma = m0 + j + half * 8;            // C/D layout: VGPR j -> M=j / j+8
            const int mb = ma + 16;
            if (ma < N_NODES) ps[(size_t)ma * NCOMB + n] = acc0[t][j];
            if (mb < N_NODES) ps[(size_t)mb * NCOMB + n] = acc1[t][j];
        }
    }
}

// =====================================================================
// Split-K reduction: hpre = sum_ks partial[ks]
// =====================================================================
__global__ __launch_bounds__(256)
void reduce_ksplit_kernel(const float* __restrict__ partial, float* __restrict__ hpre) {
    int i = blockIdx.x * blockDim.x + threadIdx.x;
    if (i < NL) {
        float s = 0.0f;
#pragma unroll 8
        for (int j = 0; j < KSPLIT; ++j) s += partial[(size_t)j * NL + i];
        hpre[i] = s;
    }
}

// =====================================================================
// Degree accumulation over edge targets (self loop pre-seeded = 1)
// =====================================================================
__global__ __launch_bounds__(256)
void degree_kernel(const long long* __restrict__ ei, int E, float* __restrict__ deg) {
    int i = blockIdx.x * blockDim.x + threadIdx.x;
    if (i < E) atomicAdd(&deg[(int)ei[E + i]], 1.0f);
}

// =====================================================================
// Scatter 1: h1acc[col] += norm * hpre_W1[row]  (edges + self loops)
// =====================================================================
__global__ __launch_bounds__(256)
void scatter1_kernel(const long long* __restrict__ ei, int E,
                     const float* __restrict__ hpre, const float* __restrict__ deg,
                     float* __restrict__ h1acc) {
    int idx  = blockIdx.x * blockDim.x + threadIdx.x;
    int item = idx >> 3;
    int f0   = (idx & 7) * 8;
    if (item < E) {
        int r = (int)ei[item], c = (int)ei[E + item];
        float norm = rsqrtf(deg[r]) * rsqrtf(deg[c]);
        const float* src = hpre + (size_t)r * NCOMB;     // cols 0..63 = x@W1
        float* dst = h1acc + (size_t)c * HDIM;
#pragma unroll
        for (int f = f0; f < f0 + 8; ++f) atomicAdd(&dst[f], src[f] * norm);
    } else if (item < E + N_NODES) {
        int n = item - E;
        float norm = 1.0f / deg[n];                      // dinv^2 (self loop)
        const float* src = hpre + (size_t)n * NCOMB;
        float* dst = h1acc + (size_t)n * HDIM;
#pragma unroll
        for (int f = f0; f < f0 + 8; ++f) atomicAdd(&dst[f], src[f] * norm);
    }
}

// =====================================================================
// SiLU(res + gcn1) with wave32-reduced global sum / sumsq
// =====================================================================
__global__ __launch_bounds__(256)
void silu_reduce_kernel(const float* __restrict__ hpre, const float* __restrict__ h1acc,
                        const float* __restrict__ b1, const float* __restrict__ LPb,
                        float* __restrict__ h, float* __restrict__ sums) {
    int i = blockIdx.x * blockDim.x + threadIdx.x;
    float s = 0.0f, s2 = 0.0f;
    if (i < NH) {
        int m = i >> 6, f = i & 63;
        float res = hpre[(size_t)m * NCOMB + 64 + f] + LPb[f];   // x@LPw + LPb
        float g   = h1acc[i] + b1[f];
        float v   = res + g;
        float hv  = v / (1.0f + __expf(-v));                     // SiLU
        h[i] = hv;
        s = hv; s2 = hv * hv;
    }
#pragma unroll
    for (int off = 16; off > 0; off >>= 1) {                     // wave32 reduce
        s  += __shfl_xor(s,  off, 32);
        s2 += __shfl_xor(s2, off, 32);
    }
    if ((threadIdx.x & 31) == 0) {
        atomicAdd(&sums[0], s);
        atomicAdd(&sums[1], s2);
    }
}

// =====================================================================
// Global LayerNorm over all 646*64 elements (no affine)
// =====================================================================
__global__ __launch_bounds__(256)
void norm_kernel(const float* __restrict__ h, const float* __restrict__ sums,
                 float* __restrict__ hn) {
    int i = blockIdx.x * blockDim.x + threadIdx.x;
    if (i < NH) {
        const float inv = 1.0f / (float)NH;
        float mu  = sums[0] * inv;
        float var = sums[1] * inv - mu * mu;
        hn[i] = (h[i] - mu) * rsqrtf(var + 1e-5f);
    }
}

// =====================================================================
// GEMM2: hpre2[646,128] = hn @ W2   (bf16 WMMA, K=64, direct store)
// grid = MGROUPS, block = 32; B from pre-transposed bf16 Wt2 [128][64].
// =====================================================================
__global__ __launch_bounds__(32)
void gemm2_wmma(const float* __restrict__ hn,
                const unsigned short* __restrict__ wt2,
                float* __restrict__ hpre2) {
    const int lane = threadIdx.x;
    const int half = lane >> 4;
    const int l15  = lane & 15;
    const int kk   = half * 8;
    const int m0   = blockIdx.x * 32;

    int mr0 = m0 + l15;      if (mr0 > N_NODES - 1) mr0 = N_NODES - 1;
    int mr1 = m0 + 16 + l15; if (mr1 > N_NODES - 1) mr1 = N_NODES - 1;
    const float* __restrict__ arow0 = hn + (size_t)mr0 * HDIM;
    const float* __restrict__ arow1 = hn + (size_t)mr1 * HDIM;
    const unsigned short* __restrict__ brow = wt2 + (size_t)l15 * HDIM;

    v8f zero = {};
    v8f acc0[NTILES], acc1[NTILES];
#pragma unroll
    for (int t = 0; t < NTILES; ++t) { acc0[t] = zero; acc1[t] = zero; }

#pragma unroll
    for (int k0 = 0; k0 < HDIM; k0 += 32) {
        v16bf A0 = load_a_bf16(arow0 + k0 + kk);
        v16bf A1 = load_a_bf16(arow1 + k0 + kk);
#pragma unroll
        for (int t = 0; t < NTILES; ++t) {
            union { v16bf v; uint4 q[2]; } B;
            const unsigned short* p = brow + (size_t)(t * 16) * HDIM + k0 + kk;
            B.q[0] = *(const uint4*)(p);
            B.q[1] = *(const uint4*)(p + 16);
            acc0[t] = __builtin_amdgcn_wmma_f32_16x16x32_bf16(
                false, A0, false, B.v, (short)0, acc0[t], false, false);
            acc1[t] = __builtin_amdgcn_wmma_f32_16x16x32_bf16(
                false, A1, false, B.v, (short)0, acc1[t], false, false);
        }
    }
#pragma unroll
    for (int t = 0; t < NTILES; ++t) {
#pragma unroll
        for (int j = 0; j < 8; ++j) {
            const int n  = t * 16 + l15;
            const int ma = m0 + j + half * 8;
            const int mb = ma + 16;
            if (ma < N_NODES) hpre2[(size_t)ma * LDIM + n] = acc0[t][j];
            if (mb < N_NODES) hpre2[(size_t)mb * LDIM + n] = acc1[t][j];
        }
    }
}

// =====================================================================
// Scatter 2: out[col] += norm * hpre2[row]  (out pre-seeded with b2)
// =====================================================================
__global__ __launch_bounds__(256)
void scatter2_kernel(const long long* __restrict__ ei, int E,
                     const float* __restrict__ hpre2, const float* __restrict__ deg,
                     float* __restrict__ out) {
    int idx  = blockIdx.x * blockDim.x + threadIdx.x;
    int item = idx >> 3;
    int f0   = (idx & 7) * 16;
    if (item < E) {
        int r = (int)ei[item], c = (int)ei[E + item];
        float norm = rsqrtf(deg[r]) * rsqrtf(deg[c]);
        const float* src = hpre2 + (size_t)r * LDIM;
        float* dst = out + (size_t)c * LDIM;
#pragma unroll
        for (int f = f0; f < f0 + 16; ++f) atomicAdd(&dst[f], src[f] * norm);
    } else if (item < E + N_NODES) {
        int n = item - E;
        float norm = 1.0f / deg[n];
        const float* src = hpre2 + (size_t)n * LDIM;
        float* dst = out + (size_t)n * LDIM;
#pragma unroll
        for (int f = f0; f < f0 + 16; ++f) atomicAdd(&dst[f], src[f] * norm);
    }
}

// =====================================================================
extern "C" void kernel_launch(void* const* d_in, const int* in_sizes, int n_in,
                              void* d_out, int out_size, void* d_ws, size_t ws_size,
                              hipStream_t stream) {
    const float*     x   = (const float*)d_in[0];
    const long long* ei  = (const long long*)d_in[1];   // int64 edge_index [2,E]
    const float*     W1  = (const float*)d_in[2];
    const float*     b1  = (const float*)d_in[3];
    const float*     W2  = (const float*)d_in[4];
    const float*     b2  = (const float*)d_in[5];
    const float*     LPw = (const float*)d_in[6];
    const float*     LPb = (const float*)d_in[7];
    float*           out = (float*)d_out;
    const int E = in_sizes[1] / 2;

    // -------- workspace layout --------
    float* ws      = (float*)d_ws;
    float* partial = ws;                                // [KSPLIT * NL] = 2,646,016 f
    float* hpre    = partial + (size_t)KSPLIT * NL;     // [NL]
    float* h1acc   = hpre  + NL;                        // [NH]
    float* h       = h1acc + NH;                        // [NH]
    float* hn      = h     + NH;                        // [NH]
    float* hpre2   = hn    + NH;                        // [NL]
    float* deg     = hpre2 + NL;                        // [648]
    float* sums    = deg + 648;                         // [8]
    unsigned short* wt1 = (unsigned short*)(sums + 8);  // [128 * FDIM] bf16 (16B-aligned)
    unsigned short* wt2 = wt1 + (size_t)NCOMB * FDIM;   // [128 * 64]  bf16

    // -------- one-time weight transposes (bf16) --------
    transpose_bf16_kernel<<<dim3(FDIM / 64, 1), 256, 0, stream>>>(W1,  wt1, 64, FDIM);
    transpose_bf16_kernel<<<dim3(FDIM / 64, 1), 256, 0, stream>>>(LPw, wt1 + (size_t)64 * FDIM, 64, FDIM);
    transpose_bf16_kernel<<<dim3(1, 2),         256, 0, stream>>>(W2,  wt2, 128, HDIM);

    init_kernel<<<(NL + 255) / 256, 256, 0, stream>>>(h1acc, deg, sums, out, b2);
    degree_kernel<<<(E + 255) / 256, 256, 0, stream>>>(ei, E, deg);

    // -------- dominant fused GEMM over x --------
    gemm1_wmma<<<dim3(MGROUPS, KSPLIT), 32, 0, stream>>>(x, wt1, partial);
    reduce_ksplit_kernel<<<(NL + 255) / 256, 256, 0, stream>>>(partial, hpre);

    int items1 = (E + N_NODES) * 8;
    scatter1_kernel<<<(items1 + 255) / 256, 256, 0, stream>>>(ei, E, hpre, deg, h1acc);

    silu_reduce_kernel<<<(NH + 255) / 256, 256, 0, stream>>>(hpre, h1acc, b1, LPb, h, sums);
    norm_kernel<<<(NH + 255) / 256, 256, 0, stream>>>(h, sums, hn);

    gemm2_wmma<<<MGROUPS, 32, 0, stream>>>(hn, wt2, hpre2);

    int items2 = (E + N_NODES) * 8;
    scatter2_kernel<<<(items2 + 255) / 256, 256, 0, stream>>>(ei, E, hpre2, deg, out);
}
